// Net_66245575573837
// MI455X (gfx1250) — matrix operand order repaired and checked by reading, not compiled, not run on hardware
//
#include <hip/hip_runtime.h>

// Problem constants (match reference)
#define NNODES   50000
#define NPAD     50048   // NNODES rounded up to a multiple of 128 (tile size)
#define NEDGES   400000
#define NGRAPHS  500
#define NHEADS   3
#define CCH      30
#define DH       10      // CCH / NHEADS
#define INCH     5
#define NLAYERS  6
#define HIST     7       // stored history layers 0..6

typedef __attribute__((ext_vector_type(16))) _Float16 v16h;
typedef __attribute__((ext_vector_type(8)))  _Float16 v8h;
typedef __attribute__((ext_vector_type(8)))  float    v8f;

// ---------------------------------------------------------------------------
// WMMA projection kernel: y = act @ W^T + b   (templated on MODE; all node
// buffers padded to NPAD rows so the store path has NO per-element guards)
//   MODE 0: h = relu(x @ W1^T + b1)       in n*5 (guarded),  out (n*7)*30
//   MODE 1: q = x_all[:,L-1] @ Wq^T + bq  in (n*7+L-1)*30,   out n*30
//   MODE 2: k/v = x_all[:,l] @ W^T + b    l = blockIdx.y,
//                                         in (n*7+l)*30,     out (n*L+l)*30
// Tile: 16 rows x K=32 (zero-padded) x 32 cols (2 x 16-wide WMMA tiles).
// One wave per 16-row tile, 8 waves / 256-thread block -> 128 rows per block.
// W is pre-swizzled into LDS in B-fragment order -> pure ds_load_b128 reads.
// ---------------------------------------------------------------------------
template <int MODE>
__global__ __launch_bounds__(256)
void proj_wmma(const float* __restrict__ act,
               const float* __restrict__ W,     // [30 x IND] row-major (out,in)
               const float* __restrict__ bias,  // [30]
               float* __restrict__ out,
               int L)
{
    constexpr int  IND    = (MODE == 0) ? INCH : CCH;  // K (zero-padded to 32)
    constexpr bool DORELU = (MODE == 0);

    // sBf[nt][lane][j] = W[col*IND + kb2+j], col = nt*16+(lane&15),
    // kb2 = (lane>>4)*16; zero outside [30 x IND].
    __shared__ __align__(32) _Float16 sBf[2 * 32 * 16];
    __shared__ float sBias[32];
    __shared__ __align__(32) _Float16 sA[8][16 * 32];  // per-wave 16x32 A tile

    const int tid  = threadIdx.x;
    const int wid  = tid >> 5;
    const int lane = tid & 31;
    const int l    = (MODE == 2) ? (int)blockIdx.y : 0;

    // Cooperative pre-swizzled weight load (1024 halves, 4 consecutive per
    // thread -> one b128 global read + one b64 LDS write each).
    {
        const int base = tid * 4;
        const int nt    = base >> 9;
        const int lane2 = (base >> 4) & 31;
        const int col   = ((nt << 4)) + (lane2 & 15);
        const int kb2   = (lane2 >> 4) << 4;
        #pragma unroll
        for (int t = 0; t < 4; ++t) {
            const int k = kb2 + (base & 15) + t;
            float w = (col < CCH && k < IND) ? W[col * IND + k] : 0.0f;
            sBf[base + t] = (_Float16)w;
        }
    }
    if (tid < 32) sBias[tid] = (tid < CCH) ? bias[tid] : 0.0f;
    __syncthreads();

    const int rowBase = blockIdx.x * 128 + wid * 16;

    // Stage 16x32 A tile, lane-major: iteration i loads one full row with
    // col = lane -> coalesced 128B per wave, conflict-free LDS store.
    #pragma unroll
    for (int i = 0; i < 16; ++i) {
        const int R = rowBase + i;
        float val = 0.0f;
        if (MODE == 0) {
            if (R < NNODES && lane < IND) val = act[(long long)R * INCH + lane];
        } else {
            // padded x_all: unconditional read; lanes 30/31 pick up finite
            // neighbor floats that multiply the zeroed B rows 30/31 -> 0.
            long long ofs = (MODE == 1)
                ? ((long long)R * HIST + (L - 1)) * CCH
                : ((long long)R * HIST + l) * CCH;
            val = act[ofs + lane];
        }
        sA[wid][(i << 5) + lane] = (_Float16)val;
    }
    // sA slab is wave-private; same-wave DS ops are in-order.

    // A fragment (16-bit A 16x32, wave32):
    //   lane<16 : row=lane,    K in {0..15}
    //   lane>=16: row=lane-16, K in {16..31}
    const int arow  = lane & 15;
    const int kbase = (lane >> 4) << 4;     // 0 or 16
    const v8h alo = *(const v8h*)&sA[wid][(arow << 5) + kbase];
    const v8h ahi = *(const v8h*)&sA[wid][(arow << 5) + kbase + 8];
    const v16h a = __builtin_shufflevector(alo, ahi,
        0, 1, 2, 3, 4, 5, 6, 7, 8, 9, 10, 11, 12, 13, 14, 15);

    #pragma unroll
    for (int nt = 0; nt < 2; ++nt) {
        const int col = nt * 16 + (lane & 15);
        const v16h b = *(const v16h*)&sBf[nt * 512 + lane * 16];

        v8f c = {};
        c = __builtin_amdgcn_wmma_f32_16x16x32_f16(
                /*neg_a=*/false, a, /*neg_b=*/false, b,
                /*c_mod=*/(short)0, c, /*reuse_a=*/false, /*reuse_b=*/false);

        // C/D layout: VGPR i -> M = i + 8*(lane>=16), N = lane&15 (+16*nt)
        if (col < CCH) {                       // uniform-per-lane guard only
            const float bcol = sBias[col];
            const int   Mb   = (lane >> 4) << 3;
            #pragma unroll
            for (int i = 0; i < 8; ++i) {
                const int R = rowBase + Mb + i;
                float y = c[i] + bcol;
                if (DORELU) y = fmaxf(y, 0.0f);
                long long ofs;
                if (MODE == 0)      ofs = (long long)R * (HIST * CCH);
                else if (MODE == 1) ofs = (long long)R * CCH;
                else                ofs = ((long long)R * L + l) * CCH;
                out[ofs + col] = y;            // unconditional: buffers padded
            }
        }
    }
}

// ---------------------------------------------------------------------------
// Edge attention: ONE thread per edge, all 3 heads. eidx/dis/q fetched once
// per edge; k/v rows each gathered once and reused by all heads. q/k/v live
// in the 192 MB L2 (<= 90 MB total), so this is L2-gather + atomic bound.
// v-rows are prefetched (global_prefetch_b8) while the k dot-products run.
// ---------------------------------------------------------------------------
__global__ __launch_bounds__(256)
void edge_attn(const long long* __restrict__ eidx,  // [2*NEDGES] (src, dst)
               const float* __restrict__ dis,
               const float* __restrict__ q,         // [NPAD,30]
               const float* __restrict__ kbuf,      // [NPAD,L,30]
               const float* __restrict__ vbuf,      // [NPAD,L,30]
               float* __restrict__ agg,             // [NPAD,30]
               int L)
{
    const int F = NEDGES + NNODES;                  // edges + self loops
    int e = blockIdx.x * blockDim.x + threadIdx.x;
    if (e >= F) return;

    int src, dst;
    if (e < NEDGES) { src = (int)eidx[e]; dst = (int)eidx[NEDGES + e]; }
    else            { src = dst = e - NEDGES; }

    const float nrm = dis[src] * dis[dst];

    float qr[CCH];
    const float* qp = q + (long long)dst * CCH;
    #pragma unroll
    for (int c = 0; c < CCH; ++c) qr[c] = qp[c];

    const float* kb = kbuf + (long long)src * L * CCH;
    const float* vb = vbuf + (long long)src * L * CCH;

    const float scale = 0.31622776601683794f;       // 1/sqrt(dh)
    float s[NHEADS][NLAYERS];
    #pragma unroll
    for (int l = 0; l < NLAYERS; ++l) {
        if (l < L) {
            const float* kp = kb + l * CCH;
            __builtin_prefetch(vb + l * CCH, 0, 1); // warm v row during k dots
            #pragma unroll
            for (int h = 0; h < NHEADS; ++h) {
                float acc = 0.0f;
                #pragma unroll
                for (int d = 0; d < DH; ++d) acc += qr[h * DH + d] * kp[h * DH + d];
                s[h][l] = acc * scale;
            }
        }
    }

    float wn[NHEADS];
    #pragma unroll
    for (int h = 0; h < NHEADS; ++h) {
        float m = -3.0e38f;
        #pragma unroll
        for (int l = 0; l < NLAYERS; ++l) if (l < L) m = fmaxf(m, s[h][l]);
        float sum = 0.0f;
        #pragma unroll
        for (int l = 0; l < NLAYERS; ++l) {
            float w = (l < L) ? __expf(s[h][l] - m) : 0.0f;
            s[h][l] = w;
            sum += w;
        }
        wn[h] = nrm / sum;
    }

    float o[CCH];
    #pragma unroll
    for (int c = 0; c < CCH; ++c) o[c] = 0.0f;
    #pragma unroll
    for (int l = 0; l < NLAYERS; ++l) {
        if (l < L) {
            const float* vp = vb + l * CCH;
            #pragma unroll
            for (int h = 0; h < NHEADS; ++h) {
                const float w = s[h][l];
                #pragma unroll
                for (int d = 0; d < DH; ++d) o[h * DH + d] += w * vp[h * DH + d];
            }
        }
    }

    float* ap = agg + (long long)dst * CCH;
    #pragma unroll
    for (int c = 0; c < CCH; ++c)
        atomicAdd(&ap[c], o[c] * wn[c / DH]);       // c/DH constant per slot
}

// ------------------------- small helper kernels ----------------------------
__global__ void init_deg(float* deg) {
    int i = blockIdx.x * blockDim.x + threadIdx.x;
    if (i < NNODES) deg[i] = 1.0f;                  // self loop
}
__global__ void accum_deg(const long long* __restrict__ eidx, float* deg) {
    int e = blockIdx.x * blockDim.x + threadIdx.x;
    if (e < NEDGES) atomicAdd(&deg[(int)eidx[NEDGES + e]], 1.0f);
}
__global__ void calc_dis(const float* __restrict__ deg, float* dis) {
    int i = blockIdx.x * blockDim.x + threadIdx.x;
    if (i < NNODES) dis[i] = rsqrtf(deg[i]);        // deg >= 1 always
}
__global__ void zero_buf(float* p, int n) {
    int i = blockIdx.x * blockDim.x + threadIdx.x;
    if (i < n) p[i] = 0.0f;
}
__global__ void relu_layer(const float* __restrict__ agg,
                           float* __restrict__ x_all, int L) {
    int i = blockIdx.x * blockDim.x + threadIdx.x;
    if (i < NNODES * CCH) {
        int n = i / CCH, c = i - n * CCH;
        x_all[((long long)n * HIST + L) * CCH + c] = fmaxf(agg[i], 0.0f);
    }
}
__global__ void pool_init(float* out, const float* __restrict__ b2) {
    int i = blockIdx.x * blockDim.x + threadIdx.x;
    if (i < NGRAPHS) out[i] = b2[0];
}
__global__ void pool_accum(const float* __restrict__ x_all,
                           const int* __restrict__ batch,
                           const float* __restrict__ W2,
                           float* out) {
    int n = blockIdx.x * blockDim.x + threadIdx.x;
    if (n < NNODES) {
        const float* xp = x_all + ((long long)n * HIST + NLAYERS) * CCH;
        float dot = 0.0f;
        #pragma unroll
        for (int c = 0; c < CCH; ++c) dot += xp[c] * W2[c];
        atomicAdd(&out[batch[n]], dot);
    }
}

// ---------------------------------------------------------------------------
extern "C" void kernel_launch(void* const* d_in, const int* in_sizes, int n_in,
                              void* d_out, int out_size, void* d_ws, size_t ws_size,
                              hipStream_t stream)
{
    const float*     x     = (const float*)d_in[0];
    const long long* eidx  = (const long long*)d_in[1];   // int64 per reference
    const int*       batch = (const int*)d_in[2];
    const float*     W1    = (const float*)d_in[3];
    const float*     b1    = (const float*)d_in[4];
    const float*     Wq    = (const float*)d_in[5];
    const float*     bq    = (const float*)d_in[6];
    const float*     Wk    = (const float*)d_in[7];
    const float*     bk    = (const float*)d_in[8];
    const float*     Wv    = (const float*)d_in[9];
    const float*     bv    = (const float*)d_in[10];
    const float*     W2    = (const float*)d_in[11];
    const float*     b2    = (const float*)d_in[12];
    float*           out   = (float*)d_out;
    (void)in_sizes; (void)n_in; (void)out_size; (void)ws_size;

    // Workspace carve-up, all node buffers padded to NPAD rows (~127 MB f32)
    float* ws    = (float*)d_ws;
    size_t o = 0;
    float* x_all = ws + o; o += (size_t)NPAD * HIST * CCH;     // [NPAD,7,30]
    float* q     = ws + o; o += (size_t)NPAD * CCH;            // [NPAD,30]
    float* kbuf  = ws + o; o += (size_t)NPAD * NLAYERS * CCH;  // [NPAD,<=6,30]
    float* vbuf  = ws + o; o += (size_t)NPAD * NLAYERS * CCH;  // [NPAD,<=6,30]
    float* agg   = ws + o; o += (size_t)NPAD * CCH;            // [NPAD,30]
    float* deg   = ws + o; o += (size_t)NPAD;
    float* dis   = ws + o; o += (size_t)NPAD;

    const int T = 256;
    const int projBlocks = NPAD / 128;   // exact: no partial tiles

    // Degrees / normalization
    init_deg <<<(NNODES + T - 1) / T, T, 0, stream>>>(deg);
    accum_deg<<<(NEDGES + T - 1) / T, T, 0, stream>>>(eidx, deg);
    calc_dis <<<(NNODES + T - 1) / T, T, 0, stream>>>(deg, dis);

    // h = relu(x @ W1^T + b1) -> x_all layer 0
    proj_wmma<0><<<projBlocks, 256, 0, stream>>>(x, W1, b1, x_all, 1);

    for (int t = 0; t < NLAYERS; ++t) {
        const int L = t + 1;
        // q projection
        proj_wmma<1><<<projBlocks, 256, 0, stream>>>(
            x_all, Wq + t * CCH * CCH, bq + t * CCH, q, L);
        // k / v projections: grid.y = history layer l
        dim3 gkv(projBlocks, L);
        proj_wmma<2><<<gkv, 256, 0, stream>>>(
            x_all, Wk + t * CCH * CCH, bk + t * CCH, kbuf, L);
        proj_wmma<2><<<gkv, 256, 0, stream>>>(
            x_all, Wv + t * CCH * CCH, bv + t * CCH, vbuf, L);

        zero_buf<<<(NNODES * CCH + T - 1) / T, T, 0, stream>>>(agg, NNODES * CCH);

        const int F = NEDGES + NNODES;
        edge_attn<<<(F + T - 1) / T, T, 0, stream>>>(
            eidx, dis, q, kbuf, vbuf, agg, L);

        relu_layer<<<(NNODES * CCH + T - 1) / T, T, 0, stream>>>(agg, x_all, L);
    }

    // pooled @ W2^T + b2  -> out[500]
    pool_init <<<(NGRAPHS + T - 1) / T, T, 0, stream>>>(out, b2);
    pool_accum<<<(NNODES + T - 1) / T, T, 0, stream>>>(x_all, batch, W2, out);
}
